// Quantizer_58213986730739
// MI455X (gfx1250) — matrix operand order repaired and checked by reading, not compiled, not run on hardware
//
#include <hip/hip_runtime.h>
#include <stdint.h>

// ---- problem constants (from reference) ----
#define DVQ 128          // codebook dim / window size
#define KVQ 2048         // codebook entries
#define MVQ (2048 * 32)  // B * NW = 65536 rows
#define CHUNK 128        // codebook entries staged in LDS per iteration
#define NCHUNK (KVQ / CHUNK)
#define LDSTRIDE (DVQ + 8)  // +16B pad per entry -> conflict-free ds_load_b128

typedef __attribute__((ext_vector_type(16))) __bf16 v16bf;
typedef __attribute__((ext_vector_type(8)))  float  v8f;
typedef __attribute__((ext_vector_type(4)))  __bf16 v4bf;

// ---------------------------------------------------------------------------
// Prologue: per-entry squared norm (f32) + bf16 copy of the codebook.
// One wave (32 lanes) per codebook row; lane handles 4 consecutive floats.
// ---------------------------------------------------------------------------
__global__ __launch_bounds__(256) void vq_prep(const float* __restrict__ cb,
                                               unsigned short* __restrict__ cbb_raw,
                                               float* __restrict__ cbn) {
  __bf16* cbb = (__bf16*)cbb_raw;
  const int wave = threadIdx.x >> 5;
  const int lane = threadIdx.x & 31;
  const int row  = blockIdx.x * 8 + wave;

  const float4 v = *(const float4*)(cb + (size_t)row * DVQ + lane * 4);
  float s = v.x * v.x + v.y * v.y + v.z * v.z + v.w * v.w;

  v4bf b;
  b[0] = (__bf16)v.x; b[1] = (__bf16)v.y; b[2] = (__bf16)v.z; b[3] = (__bf16)v.w;
  *(v4bf*)(cbb + (size_t)row * DVQ + lane * 4) = b;

#pragma unroll
  for (int off = 16; off >= 1; off >>= 1) s += __shfl_xor(s, off);
  if (lane == 0) cbn[row] = s;
}

// ---------------------------------------------------------------------------
// Async-stage one 128-entry codebook chunk (bf16, padded stride) into LDS.
// 32 KB per chunk = 2048 x 16B transfers; each of 256 threads issues 8
// global_load_async_to_lds_b128 (tracked by ASYNCcnt, overlaps with WMMA).
// ---------------------------------------------------------------------------
__device__ __forceinline__ void stage_chunk(const __bf16* __restrict__ cbb,
                                            __bf16* dstLds, int chunk, int tid) {
  const __bf16* src = cbb + (size_t)chunk * CHUNK * DVQ;
#pragma unroll
  for (int i = 0; i < 8; ++i) {
    const int t   = tid + i * 256;   // 0..2047
    const int e   = t >> 4;          // entry within chunk
    const int seg = t & 15;          // 16-byte segment within entry
    const unsigned long long g =
        (unsigned long long)(uintptr_t)(src + (size_t)e * DVQ + seg * 8);
    const unsigned ldsOff =
        (unsigned)(uintptr_t)(dstLds + e * LDSTRIDE + seg * 8);
    asm volatile("global_load_async_to_lds_b128 %0, %1, off"
                 :: "v"(ldsOff), "v"(g) : "memory");
  }
}

// ---------------------------------------------------------------------------
// Main kernel: each wave computes argmin_k || z_row - c_k ||^2 for 16 rows
// via bf16 WMMA (score = ||c||^2 - 2 z.c), then gathers the f32 winner rows.
// Block = 8 waves = 128 rows; grid = 512 blocks.
// ---------------------------------------------------------------------------
__global__ __launch_bounds__(256) void vq_main(const float* __restrict__ ze,
                                               const float* __restrict__ cb,
                                               const unsigned short* __restrict__ cbb_raw,
                                               const float* __restrict__ cbn,
                                               float* __restrict__ out) {
  const __bf16* cbb = (const __bf16*)cbb_raw;
  __shared__ __bf16 sB[2][CHUNK * LDSTRIDE];  // double-buffered, ~68 KB

  const int wave = threadIdx.x >> 5;
  const int lane = threadIdx.x & 31;
  const int hf   = lane >> 4;    // half-wave id
  const int m    = lane & 15;    // A: row in tile / B,C: column in tile
  const int rowBase = blockIdx.x * 128 + wave * 16;

  // ---- A fragments: 16 rows x 128 d, bf16, kept in VGPRs all K-loop ----
  // 16-bit A 16x32 layout: lane half hf holds k = j*32 + hf*16 + {0..15} in
  // elems 0..15 (contiguous, matching B).
  v16bf a[4];
  const float* zrow = ze + (size_t)(rowBase + m) * DVQ;
#pragma unroll
  for (int j = 0; j < 4; ++j) {
    const int d0 = j * 32 + hf * 16;
    const float* p = zrow + d0;
    const float* q = zrow + d0 + 8;
#pragma unroll
    for (int e = 0; e < 8; ++e) a[j][e]     = (__bf16)p[e];
#pragma unroll
    for (int e = 0; e < 8; ++e) a[j][8 + e] = (__bf16)q[e];
  }

  // ---- running (min, argmin); per lane: 8 row-slots of the C layout ----
  float minv[8];
  int   mini[8];
#pragma unroll
  for (int v = 0; v < 8; ++v) { minv[v] = 3.4e38f; mini[v] = 0; }

  stage_chunk(cbb, sB[0], 0, threadIdx.x);
  asm volatile("s_wait_asynccnt 0" ::: "memory");
  __syncthreads();

  for (int chunk = 0; chunk < NCHUNK; ++chunk) {
    const int cur = chunk & 1;
    if (chunk + 1 < NCHUNK)
      stage_chunk(cbb, sB[cur ^ 1], chunk + 1, threadIdx.x);  // overlap DMA

    const __bf16* Bc = sB[cur];
#pragma unroll
    for (int nt = 0; nt < 8; ++nt) {  // 8 tiles of 16 codebook entries
      v8f c = {};
#pragma unroll
      for (int j = 0; j < 4; ++j) {   // accumulate over D = 4 x 32
        // 16-bit B 32x16: lane = column n, half hf holds k = j*32+hf*16..+15
        const v16bf b = *(const v16bf*)(Bc + (nt * 16 + m) * LDSTRIDE +
                                        j * 32 + hf * 16);
        c = __builtin_amdgcn_wmma_f32_16x16x32_bf16(
                false, a[j], false, b, (short)0, c, false, false);
      }
      const int kglob = chunk * CHUNK + nt * 16 + m;
      const float cn = cbn[kglob];
#pragma unroll
      for (int v = 0; v < 8; ++v) {   // score = ||c||^2 - 2 z.c
        const float s = fmaf(-2.0f, c[v], cn);
        const bool p = s < minv[v];   // strict < keeps lowest k on ties
        minv[v] = p ? s : minv[v];
        mini[v] = p ? kglob : mini[v];
      }
    }
    asm volatile("s_wait_asynccnt 0" ::: "memory");
    __syncthreads();  // next chunk's buffer fully landed, readers done
  }

  // ---- cross-lane argmin within each 16-lane half (C layout: lane = N) ----
#pragma unroll
  for (int off = 1; off < 16; off <<= 1) {
#pragma unroll
    for (int v = 0; v < 8; ++v) {
      const float ov = __shfl_xor(minv[v], off);
      const int   oi = __shfl_xor(mini[v], off);
      const bool p = (ov < minv[v]) || (ov == minv[v] && oi < mini[v]);
      minv[v] = p ? ov : minv[v];
      mini[v] = p ? oi : mini[v];
    }
  }

  // ---- gather winning codebook rows (f32, bit-exact output) ----
  // lanes 0-15 hold rows 0..7 (vgpr v -> row v), lanes 16-31 rows 8..15.
#pragma unroll
  for (int r = 0; r < 16; ++r) {
    const int idx = __shfl(mini[r & 7], (r < 8) ? 0 : 16);
    const float4* srcRow = (const float4*)(cb + (size_t)idx * DVQ);
    float4* dstRow = (float4*)(out + (size_t)(rowBase + r) * DVQ);
    dstRow[lane] = srcRow[lane];  // 32 lanes x 16B = 512B row, coalesced
  }
}

// ---------------------------------------------------------------------------
extern "C" void kernel_launch(void* const* d_in, const int* in_sizes, int n_in,
                              void* d_out, int out_size, void* d_ws, size_t ws_size,
                              hipStream_t stream) {
  const float* ze = (const float*)d_in[0];   // [65536, 128] f32
  const float* cb = (const float*)d_in[1];   // [2048, 128] f32
  unsigned short* cbb = (unsigned short*)d_ws;                       // bf16 codebook (512 KB)
  float* cbn = (float*)((char*)d_ws + (size_t)KVQ * DVQ * sizeof(unsigned short)); // norms (8 KB)
  float* out = (float*)d_out;

  vq_prep<<<KVQ / 8, 256, 0, stream>>>(cb, cbb, cbn);
  vq_main<<<MVQ / 128, 256, 0, stream>>>(ze, cb, cbb, cbn, out);
}
